// RNNEncoderHat_70351564309220
// MI455X (gfx1250) — matrix operand order repaired and checked by reading, not compiled, not run on hardware
//
#include <hip/hip_runtime.h>
#include <hip/hip_bf16.h>
#include <math.h>

// Problem constants (reference: B,S,I,H,E = 64,512,256,512,512)
#define Bn 64
#define Sn 512
#define In 256
#define Hn 512
#define En 512
#define ROWP 520   // LDS row pitch in bf16 elems: 260 dwords == 4 (mod 64 banks)

typedef __bf16 bf16;
typedef __attribute__((ext_vector_type(16))) __bf16 v16bf;
typedef __attribute__((ext_vector_type(8)))  float  v8f;

union Frag32 { v16bf v; uint4 q[2]; };        // A/B operand fragment (32 B/lane)
union Frag16 { uint4 q; bf16 e[8]; };         // C/D bf16 fragment (16 B/lane)

// ---------------------------------------------------------------------------
// Pack an (Hout x K) f32 weight matrix (used as out = in @ W^T) into bf16
// WMMA B-fragment order. For n-tile nt (16 cols of out) and k-slice kk (32):
//   frag[(nt*KK + kk)*512 + L*16 + e] = W[nt*16 + (L&15)][kk*32 + k(L>>4, e)]
// B layout (16-bit 32x16): k(h,e) = 8h+e for e<8, 16+8h+(e-8) for e>=8,
// column N = lane&15.
// ---------------------------------------------------------------------------
__global__ void pack_weight_kernel(const float* __restrict__ W, bf16* __restrict__ out,
                                   int K, int KK, int total) {
  int idx = blockIdx.x * blockDim.x + threadIdx.x;
  if (idx >= total) return;
  int e  = idx & 15;
  int L  = (idx >> 4) & 31;
  int t  = idx >> 9;
  int kk = t % KK;
  int nt = t / KK;
  int row = nt * 16 + (L & 15);
  int col = kk * 32 + (L >> 4) * 8 + (e & 7) + (e >> 3) * 16;
  out[idx] = (bf16)W[row * K + col];
}

// x[b][s][i] f32  ->  xbf[s][b][i] bf16   (rows of the phase-1 GEMM are (s*B+b))
__global__ void convert_x_kernel(const float* __restrict__ x, bf16* __restrict__ xbf) {
  int idx = blockIdx.x * blockDim.x + threadIdx.x;   // over Sn*Bn*In = 2^23
  int i = idx & (In - 1);
  int b = (idx >> 8) & (Bn - 1);
  int s = idx >> 14;
  xbf[idx] = (bf16)x[(b * Sn + s) * In + i];
}

// ---------------------------------------------------------------------------
// Phase 1: xw = x @ W_ih^T + b_ih + b_hh, stored bf16 in WMMA-D FRAGMENT ORDER:
//   xwf[((mt*32 + nt)*256) + lane*8 + r]  (one 512B tile per 16x16 D tile)
// so the scan kernel C-initializes with a single coalesced b128 load per lane.
// One wave per 16x16 output tile; K = I = 256 -> 8 wmma per wave.
// ---------------------------------------------------------------------------
__global__ void gemm_xw_kernel(const bf16* __restrict__ xbf,
                               const bf16* __restrict__ wih_frag,
                               const float* __restrict__ b_ih,
                               const float* __restrict__ b_hh,
                               bf16* __restrict__ xwf) {
  const int lane = threadIdx.x & 31;
  const int wv   = threadIdx.x >> 5;
  const int gw   = blockIdx.x * (blockDim.x >> 5) + wv;
  const int nt   = gw & 31;        // 32 n-tiles over H
  const int mt   = gw >> 5;        // 2048 m-tiles over S*B
  const int m    = lane & 15;
  const int half = lane >> 4;
  const int row  = mt * 16 + m;

  v8f acc = {};
  for (int kk = 0; kk < In / 32; ++kk) {
    Frag32 a, b;
    const bf16* ap = xbf + row * In + kk * 32 + half * 8;
    a.q[0] = *(const uint4*)ap;
    a.q[1] = *(const uint4*)(ap + 16);
    const bf16* bp = wih_frag + (nt * (In / 32) + kk) * 512 + lane * 16;
    b.q[0] = *(const uint4*)bp;
    b.q[1] = *(const uint4*)(bp + 8);
    acc = __builtin_amdgcn_wmma_f32_16x16x32_bf16(false, a.v, false, b.v,
                                                  (short)0, acc, false, false);
  }
  const float bias = b_ih[nt * 16 + m] + b_hh[nt * 16 + m];  // D: N = lane&15
  Frag16 f;
  #pragma unroll
  for (int r = 0; r < 8; ++r) f.e[r] = (bf16)(acc[r] + bias);
  *(uint4*)(xwf + (size_t)(mt * 32 + nt) * 256 + lane * 8) = f.q;
}

// ---------------------------------------------------------------------------
// Phase 2: sequential scan. 4 persistent WGs (one per 16-batch-row tile),
// 32 waves each (one per 16-col tile of H==E). h/a state ping-ponged in LDS
// (bank-conflict-free padded pitch); weights streamed from L2 in packed
// B-fragment order; xw read as 512B fragment tiles.
//   h_t = tanh(xw_t + h_{t-1} @ Whh^T)
//   a_t = tanh(h_t @ Wya^T + a_{t-1} @ Waa^T + b_ya + b_aa)
// ---------------------------------------------------------------------------
__global__ __launch_bounds__(1024, 1)
void rnn_scan_kernel(const bf16* __restrict__ xwf,
                     const bf16* __restrict__ whh_frag,
                     const bf16* __restrict__ wya_frag,
                     const bf16* __restrict__ waa_frag,
                     const float* __restrict__ b_ya,
                     const float* __restrict__ b_aa,
                     float* __restrict__ out) {
  __shared__ bf16 hbuf[2][16][ROWP];   // ~32.5 KB
  __shared__ bf16 abuf[2][16][ROWP];   // ~32.5 KB

  const int tid  = threadIdx.x;
  const int lane = tid & 31;
  const int wv   = tid >> 5;         // 0..31: n-tile
  const int m    = lane & 15;
  const int half = lane >> 4;
  const int bt   = blockIdx.x;       // batch tile
  const int b0   = bt * 16;
  const int h0   = wv * 16;

  for (int i = tid; i < 16 * ROWP; i += 1024) {
    (&hbuf[0][0][0])[i] = (bf16)0.f;
    (&abuf[0][0][0])[i] = (bf16)0.f;
  }
  __syncthreads();

  const float bias_a = b_ya[h0 + m] + b_aa[h0 + m];

  for (int s = 0; s < Sn; ++s) {
    const int cur = s & 1, nxt = cur ^ 1;

    // ---- h update: acc C-initialized with the xw fragment tile ----
    const size_t tbase = (size_t)((s * 4 + bt) * 32 + wv) * 256;
    Frag16 cf;
    cf.q = *(const uint4*)(xwf + tbase + lane * 8);
    v8f acc;
    #pragma unroll
    for (int r = 0; r < 8; ++r) acc[r] = (float)cf.e[r];

    for (int kk = 0; kk < Hn / 32; ++kk) {
      Frag32 a, b;
      const bf16* ap = &hbuf[cur][m][kk * 32 + half * 8];
      a.q[0] = *(const uint4*)ap;
      a.q[1] = *(const uint4*)(ap + 16);
      const bf16* bp = whh_frag + (wv * (Hn / 32) + kk) * 512 + lane * 16;
      b.q[0] = *(const uint4*)bp;
      b.q[1] = *(const uint4*)(bp + 8);
      acc = __builtin_amdgcn_wmma_f32_16x16x32_bf16(false, a.v, false, b.v,
                                                    (short)0, acc, false, false);
    }
    #pragma unroll
    for (int r = 0; r < 8; ++r)
      hbuf[nxt][r + half * 8][h0 + m] = (bf16)tanhf(acc[r]);

    if (s + 1 < Sn)  // warm the next step's 512B xw tile (global_prefetch_b8)
      __builtin_prefetch(xwf + tbase + (size_t)4 * 32 * 256 + lane * 8, 0, 1);
    __syncthreads();

    // ---- a update ----
    v8f acc2;
    #pragma unroll
    for (int r = 0; r < 8; ++r) acc2[r] = bias_a;

    for (int kk = 0; kk < Hn / 32; ++kk) {
      Frag32 a, b;
      const bf16* ap = &hbuf[nxt][m][kk * 32 + half * 8];
      a.q[0] = *(const uint4*)ap;
      a.q[1] = *(const uint4*)(ap + 16);
      const bf16* bp = wya_frag + (wv * (Hn / 32) + kk) * 512 + lane * 16;
      b.q[0] = *(const uint4*)bp;
      b.q[1] = *(const uint4*)(bp + 8);
      acc2 = __builtin_amdgcn_wmma_f32_16x16x32_bf16(false, a.v, false, b.v,
                                                     (short)0, acc2, false, false);
    }
    for (int kk = 0; kk < En / 32; ++kk) {
      Frag32 a, b;
      const bf16* ap = &abuf[cur][m][kk * 32 + half * 8];
      a.q[0] = *(const uint4*)ap;
      a.q[1] = *(const uint4*)(ap + 16);
      const bf16* bp = waa_frag + (wv * (En / 32) + kk) * 512 + lane * 16;
      b.q[0] = *(const uint4*)bp;
      b.q[1] = *(const uint4*)(bp + 8);
      acc2 = __builtin_amdgcn_wmma_f32_16x16x32_bf16(false, a.v, false, b.v,
                                                     (short)0, acc2, false, false);
    }

    if (s == Sn - 1) {
      #pragma unroll
      for (int r = 0; r < 8; ++r)
        out[(b0 + r + half * 8) * En + h0 + m] = tanhf(acc2[r]);
    } else {
      #pragma unroll
      for (int r = 0; r < 8; ++r)
        abuf[nxt][r + half * 8][h0 + m] = (bf16)tanhf(acc2[r]);
      __syncthreads();
    }
  }
}

// ---------------------------------------------------------------------------
extern "C" void kernel_launch(void* const* d_in, const int* in_sizes, int n_in,
                              void* d_out, int out_size, void* d_ws, size_t ws_size,
                              hipStream_t stream) {
  (void)in_sizes; (void)n_in; (void)out_size; (void)ws_size;
  const float* x    = (const float*)d_in[0];
  const float* W_ih = (const float*)d_in[1];
  const float* W_hh = (const float*)d_in[2];
  const float* b_ih = (const float*)d_in[3];
  const float* b_hh = (const float*)d_in[4];
  const float* W_ya = (const float*)d_in[5];
  const float* b_ya = (const float*)d_in[6];
  const float* W_aa = (const float*)d_in[7];
  const float* b_aa = (const float*)d_in[8];
  float* out = (float*)d_out;

  char* ws = (char*)d_ws;
  bf16* wih_f = (bf16*)ws; ws += (size_t)Hn * In * 2;          // 256 KB
  bf16* whh_f = (bf16*)ws; ws += (size_t)Hn * Hn * 2;          // 512 KB
  bf16* wya_f = (bf16*)ws; ws += (size_t)En * Hn * 2;          // 512 KB
  bf16* waa_f = (bf16*)ws; ws += (size_t)En * En * 2;          // 512 KB
  bf16* xbf   = (bf16*)ws; ws += (size_t)Sn * Bn * In * 2;     // 16 MB
  bf16* xwf   = (bf16*)ws;                                     // 32 MB (fragment order)

  pack_weight_kernel<<<(Hn * In) / 256, 256, 0, stream>>>(W_ih, wih_f, In, In / 32, Hn * In);
  pack_weight_kernel<<<(Hn * Hn) / 256, 256, 0, stream>>>(W_hh, whh_f, Hn, Hn / 32, Hn * Hn);
  pack_weight_kernel<<<(En * Hn) / 256, 256, 0, stream>>>(W_ya, wya_f, Hn, Hn / 32, En * Hn);
  pack_weight_kernel<<<(En * En) / 256, 256, 0, stream>>>(W_aa, waa_f, En, En / 32, En * En);

  convert_x_kernel<<<(Sn * Bn * In) / 256, 256, 0, stream>>>(x, xbf);

  // 2048 m-tiles * 32 n-tiles = 65536 wave-tiles; 8 waves / block
  gemm_xw_kernel<<<65536 / 8, 256, 0, stream>>>(xbf, wih_f, b_ih, b_hh, xwf);

  rnn_scan_kernel<<<Bn / 16, 1024, 0, stream>>>(xwf, whh_f, wya_f, waa_f, b_ya, b_aa, out);
}
